// SplitPoint_19473381720484
// MI455X (gfx1250) — matrix operand order repaired and checked by reading, not compiled, not run on hardware
//
#include <hip/hip_runtime.h>
#include <hip/hip_bf16.h>

typedef __attribute__((ext_vector_type(16))) _Float16 v16h;
typedef __attribute__((ext_vector_type(8)))  float    v8f;
typedef unsigned int u32;

constexpr int BSZ   = 8;
constexpr int CCH   = 64;
constexpr int NPTS  = 131072;
constexpr int NHALF = NPTS / 2;

constexpr int RS_TILE = 1024;      // elements per radix tile
constexpr int RS_TPB  = 256;       // threads per block
constexpr int RS_EPT  = RS_TILE / RS_TPB;   // 4 elements per thread
constexpr int NTILES  = NPTS / RS_TILE;     // 128 tiles per batch

// ---------------------------------------------------------------------------
// Kernel 1: BN+ReLU + conv1x1 channel contraction via WMMA, sigmoid, emit
// split value + descending-sortable radix key + original index.
// Each wave handles 16 points; A = conv_w replicated over the 16 rows so
// every row of D is score[n]; two 16x16x32 f16 WMMAs cover C=64.
// ---------------------------------------------------------------------------
__global__ void score_split_kernel(const float* __restrict__ x,
                                   const float* __restrict__ gamma,
                                   const float* __restrict__ beta,
                                   const float* __restrict__ rmean,
                                   const float* __restrict__ rvar,
                                   const float* __restrict__ convw,
                                   const float* __restrict__ convb,
                                   float* __restrict__ split,
                                   u32* __restrict__ keys,
                                   u32* __restrict__ idx) {
    __shared__ float sSc[CCH], sSh[CCH], sW[CCH];
    int t = threadIdx.x;
    if (t < CCH) {
        float s = gamma[t] * rsqrtf(rvar[t] + 1e-5f);
        sSc[t] = s;
        sSh[t] = beta[t] - rmean[t] * s;
        sW[t]  = convw[t];
    }
    __syncthreads();

    int lane = t & 31;
    int wave = t >> 5;
    int b    = blockIdx.y;
    int n0   = (blockIdx.x * 8 + wave) * 16;
    int hi   = lane >> 4;             // which 16-lane half of the wave
    int n    = n0 + (lane & 15);      // this lane's point (B/D column)

    // A-matrix (16x32 f16, rows replicated = conv_w chunk).
    // 16-bit A layout: lanes 0-15 halves 0..7 = K 0..7? -> per ISA table:
    // halves 0..7 = K (hi*8 .. hi*8+7), halves 8..15 = K (16+hi*8 ..).
    v16h a0, a1;
#pragma unroll
    for (int j = 0; j < 8; ++j) {
        a0[j]     = (_Float16)sW[hi * 8 + j];
        a0[8 + j] = (_Float16)sW[16 + hi * 8 + j];
        a1[j]     = (_Float16)sW[32 + hi * 8 + j];
        a1[8 + j] = (_Float16)sW[48 + hi * 8 + j];
    }

    // B-matrix (32x16 f16): lane column n, halves j<8 = channel hi*8+j,
    // halves j>=8 = channel 16+hi*8+(j-8). Apply BN scale/shift + ReLU on the fly.
    const float* xb = x + (size_t)b * CCH * NPTS;
    v16h b0, b1;
#pragma unroll
    for (int j = 0; j < 16; ++j) {
        int c0 = (j < 8) ? (hi * 8 + j) : (16 + hi * 8 + (j - 8));          // channels 0..31
        int c1 = 32 + ((j < 8) ? (hi * 8 + j) : (16 + hi * 8 + (j - 8)));     // channels 32..63
        float h0 = fmaf(xb[(size_t)c0 * NPTS + n], sSc[c0], sSh[c0]);
        float h1 = fmaf(xb[(size_t)c1 * NPTS + n], sSc[c1], sSh[c1]);
        b0[j] = (_Float16)fmaxf(h0, 0.0f);
        b1[j] = (_Float16)fmaxf(h1, 0.0f);
    }

    v8f acc = {};
    acc = __builtin_amdgcn_wmma_f32_16x16x32_f16(false, a0, false, b0,
                                                 (short)0, acc, false, false);
    acc = __builtin_amdgcn_wmma_f32_16x16x32_f16(false, a1, false, b1,
                                                 (short)0, acc, false, false);

    if (lane < 16) {
        // acc[0]: lanes 0-15 give D[M=0][N=lane] = score of point n.
        float score = acc[0] + convb[0];
        float sg = 1.0f / (1.0f + __expf(-score));
        size_t g = (size_t)b * NPTS + n;
        split[g] = sg;
        u32 u = __float_as_uint(sg);
        u32 s = (u & 0x80000000u) ? ~u : (u | 0x80000000u);  // ascending map
        keys[g] = ~s;                                        // descending key
        idx[g]  = (u32)n;
    }
}

// ---------------------------------------------------------------------------
// Radix sort pass kernels (4-bit digit, stable, deterministic)
// ---------------------------------------------------------------------------
__global__ void radix_hist(const u32* __restrict__ kin,
                           u32* __restrict__ hist, int shift) {
    __shared__ u32 h[16];
    int t = threadIdx.x;
    int tile = blockIdx.x, b = blockIdx.y;
    if (t < 16) h[t] = 0;
    __syncthreads();
    size_t base = (size_t)b * NPTS + (size_t)tile * RS_TILE;
#pragma unroll
    for (int e = 0; e < RS_EPT; ++e) {
        u32 k = kin[base + e * RS_TPB + t];
        atomicAdd(&h[(k >> shift) & 15u], 1u);
    }
    __syncthreads();
    if (t < 16) hist[((size_t)b * 16 + t) * NTILES + tile] = h[t];
}

// One block per batch: exclusive scan of 16*NTILES = 2048 counts (digit-major).
__global__ void radix_scan(u32* __restrict__ hist) {
    __shared__ u32 partial[RS_TPB];
    int t = threadIdx.x;
    int b = blockIdx.x;
    u32* hb = hist + (size_t)b * 16 * NTILES;
    constexpr int PER = (16 * NTILES) / RS_TPB;   // 8
    u32 v[PER];
    u32 sum = 0;
#pragma unroll
    for (int i = 0; i < PER; ++i) { v[i] = hb[t * PER + i]; sum += v[i]; }
    partial[t] = sum;
    __syncthreads();
    for (int off = 1; off < RS_TPB; off <<= 1) {
        u32 val = (t >= off) ? partial[t - off] : 0u;
        __syncthreads();
        partial[t] += val;
        __syncthreads();
    }
    u32 run = partial[t] - sum;    // exclusive base of this thread's segment
#pragma unroll
    for (int i = 0; i < PER; ++i) { hb[t * PER + i] = run; run += v[i]; }
}

__global__ void radix_scatter(const u32* __restrict__ kin,
                              const u32* __restrict__ iin,
                              u32* __restrict__ kout,
                              u32* __restrict__ iout,
                              const u32* __restrict__ hist, int shift) {
    __shared__ u32 cnt[16 * RS_TPB];   // digit-major: cnt[d*RS_TPB + t]
    __shared__ u32 partial[RS_TPB];
    int t = threadIdx.x;
    int tile = blockIdx.x, b = blockIdx.y;
    size_t base = (size_t)b * NPTS + (size_t)tile * RS_TILE;

#pragma unroll
    for (int d = 0; d < 16; ++d) cnt[d * RS_TPB + t] = 0;
    __syncthreads();

    // Thread t owns elements [t*4, t*4+4) in memory order -> stable ranks.
    u32 k[RS_EPT], id[RS_EPT], dg[RS_EPT], lr[RS_EPT];
#pragma unroll
    for (int e = 0; e < RS_EPT; ++e) {
        size_t g = base + t * RS_EPT + e;
        k[e]  = kin[g];
        id[e] = iin[g];
        dg[e] = (k[e] >> shift) & 15u;
        lr[e] = cnt[dg[e] * RS_TPB + t]++;   // own column: no race
    }
    __syncthreads();

    // Exclusive scan of the flat 4096-entry (digit-major) count array.
    u32 sum = 0;
#pragma unroll
    for (int i = 0; i < 16; ++i) sum += cnt[t * 16 + i];
    partial[t] = sum;
    __syncthreads();
    for (int off = 1; off < RS_TPB; off <<= 1) {
        u32 val = (t >= off) ? partial[t - off] : 0u;
        __syncthreads();
        partial[t] += val;
        __syncthreads();
    }
    u32 run = partial[t] - sum;
#pragma unroll
    for (int i = 0; i < 16; ++i) {
        u32 c = cnt[t * 16 + i];
        cnt[t * 16 + i] = run;    // now holds exclusive scan value
        run += c;
    }
    __syncthreads();

    const u32* hb = hist + (size_t)b * 16 * NTILES;
#pragma unroll
    for (int e = 0; e < RS_EPT; ++e) {
        u32 d = dg[e];
        // within-tile stable rank = (scan at (d,t)) - (scan at digit start)
        u32 within = cnt[d * RS_TPB + t] - cnt[d * RS_TPB] + lr[e];
        u32 pos = hb[d * NTILES + tile] + within;
        kout[(size_t)b * NPTS + pos] = k[e];
        iout[(size_t)b * NPTS + pos] = id[e];
    }
}

// ---------------------------------------------------------------------------
// Kernel 3: gather top-half columns of [x ; split] in sorted (descending) order
// ---------------------------------------------------------------------------
__global__ void gather_kernel(const float* __restrict__ x,
                              const float* __restrict__ split,
                              const u32* __restrict__ sidx,
                              float* __restrict__ out) {
    int b = blockIdx.y;
    int j = blockIdx.x * RS_TPB + threadIdx.x;
    u32 src = sidx[(size_t)b * NPTS + j];
    const float* xb = x + (size_t)b * CCH * NPTS;
    float* ob = out + (size_t)b * (CCH + 1) * NHALF;
    ob[(size_t)CCH * NHALF + j] = split[(size_t)b * NPTS + src];
#pragma unroll 4
    for (int ch = 0; ch < CCH; ++ch)
        ob[(size_t)ch * NHALF + j] = xb[(size_t)ch * NPTS + src];
}

// ---------------------------------------------------------------------------
extern "C" void kernel_launch(void* const* d_in, const int* in_sizes, int n_in,
                              void* d_out, int out_size, void* d_ws, size_t ws_size,
                              hipStream_t stream) {
    (void)in_sizes; (void)n_in; (void)out_size; (void)ws_size;
    const float* x     = (const float*)d_in[0];
    const float* gamma = (const float*)d_in[1];
    const float* beta  = (const float*)d_in[2];
    const float* rmean = (const float*)d_in[3];
    const float* rvar  = (const float*)d_in[4];
    const float* convw = (const float*)d_in[5];
    const float* convb = (const float*)d_in[6];
    float* out = (float*)d_out;

    char* ws = (char*)d_ws;
    const size_t ARR = (size_t)BSZ * NPTS * sizeof(u32);   // 4 MiB
    float* split = (float*)(ws);
    u32*   keyA  = (u32*)(ws + 1 * ARR);
    u32*   idxA  = (u32*)(ws + 2 * ARR);
    u32*   keyB  = (u32*)(ws + 3 * ARR);
    u32*   idxB  = (u32*)(ws + 4 * ARR);
    u32*   hist  = (u32*)(ws + 5 * ARR);   // BSZ*16*NTILES*4 = 64 KiB

    // 1) scores / split / sort keys   (wave = 16 points, 8 waves per block)
    score_split_kernel<<<dim3(NPTS / 128, BSZ), 256, 0, stream>>>(
        x, gamma, beta, rmean, rvar, convw, convb, split, keyA, idxA);

    // 2) stable LSD radix sort, 8 passes of 4 bits (even -> ends in A)
    u32 *kc = keyA, *ic = idxA, *kn = keyB, *in = idxB;
    for (int pass = 0; pass < 8; ++pass) {
        int shift = pass * 4;
        radix_hist<<<dim3(NTILES, BSZ), RS_TPB, 0, stream>>>(kc, hist, shift);
        radix_scan<<<dim3(BSZ), RS_TPB, 0, stream>>>(hist);
        radix_scatter<<<dim3(NTILES, BSZ), RS_TPB, 0, stream>>>(kc, ic, kn, in,
                                                                hist, shift);
        u32* tk = kc; kc = kn; kn = tk;
        u32* ti = ic; ic = in; in = ti;
    }

    // 3) gather top half of [x ; split] (kc/ic == A buffers after 8 passes)
    gather_kernel<<<dim3(NHALF / RS_TPB, BSZ), RS_TPB, 0, stream>>>(
        x, split, ic, out);
}
